// Attention_90615220011083
// MI455X (gfx1250) — matrix operand order repaired and checked by reading, not compiled, not run on hardware
//
#include <hip/hip_runtime.h>
#include <hip/hip_bf16.h>

// Problem constants (reference: B=2, L=2048, D=1024, H=16, DK=64)
#define B_  2
#define L_  2048
#define D_  1024
#define H_  16
#define DK_ 64
#define M_  (B_ * L_)   // 4096 rows for the projection GEMMs

typedef __bf16 bf16_t;
typedef bf16_t v16bf __attribute__((ext_vector_type(16)));
typedef bf16_t v8bf  __attribute__((ext_vector_type(8)));
typedef bf16_t v4bf  __attribute__((ext_vector_type(4)));
typedef float  v8f   __attribute__((ext_vector_type(8)));

// ---------------------------------------------------------------------------
// Optional Tensor Data Mover path (guarded so either toolchain compiles).
// ---------------------------------------------------------------------------
#if defined(__has_builtin)
#if __has_builtin(__builtin_amdgcn_tensor_load_to_lds) && \
    __has_builtin(__builtin_amdgcn_s_wait_tensorcnt)
#define USE_TDM 1
#endif
#endif
#ifndef USE_TDM
#define USE_TDM 0
#endif

#if USE_TDM
typedef unsigned int u32x4 __attribute__((ext_vector_type(4)));
typedef int          i32x4 __attribute__((ext_vector_type(4)));
typedef int          i32x8 __attribute__((ext_vector_type(8)));

// TDM load of a 128x32-element bf16 tile (row stride = 1024 elements) into
// LDS, with 16B padding appended after each 64B row so LDS rows are 80B
// (matches As rows of 40 bf16).  D# layout per CDNA5 ISA ch.8.
static __device__ __forceinline__ void tdm_load_tile_a(const bf16_t* gsrc,
                                                       unsigned lds_byte_addr) {
  const unsigned long long ga = (unsigned long long)(uintptr_t)gsrc;
  u32x4 g0;
  g0[0] = 1u;                                   // count=1, is_restore=0
  g0[1] = lds_byte_addr;                        // lds_addr
  g0[2] = (unsigned)(ga & 0xFFFFFFFFu);         // global_addr[31:0]
  g0[3] = (unsigned)((ga >> 32) & 0x01FFFFFFu)  // global_addr[56:32]
        | (2u << 30);                           // type = 2 ("image")
  i32x8 g1;
  g1[0] = (int)((1u << 16)     // data_size = 1 -> 2 bytes
              | (1u << 20)     // pad_enable
              | (3u << 22)     // pad_interval = 3 -> every 16 DWORDs (64B)
              | (3u << 25));   // pad_amount   = 3 -> 4 DWORDs (16B)
  g1[1] = (int)(32u << 16);    // tensor_dim0 = 32 (low 16 bits -> w1[31:16])
  g1[2] = (int)(128u << 16);   // tensor_dim0 hi = 0 | tensor_dim1 = 128
  g1[3] = (int)(32u << 16);    // tensor_dim1 hi = 0 | tile_dim0 = 32
  g1[4] = 128;                 // tile_dim1 = 128 | tile_dim2 = 0
  g1[5] = 1024;                // tensor_dim0_stride[31:0] = 1024 elements
  g1[6] = 0;                   // stride0 hi | stride1 lo
  g1[7] = 0;                   // stride1 hi
  const i32x4 gz = {0, 0, 0, 0};
#if __clang_major__ >= 23
  const i32x8 gz8 = {0, 0, 0, 0, 0, 0, 0, 0};
  __builtin_amdgcn_tensor_load_to_lds(g0, g1, gz, gz, gz8, 0);
#else
  __builtin_amdgcn_tensor_load_to_lds(g0, g1, gz, gz, 0);
#endif
}
#endif  // USE_TDM

// ---------------------------------------------------------------------------
// Fragment helpers (wave32 WMMA; A/B k order is this file's own, C/D per CDNA5 ISA 7.12.2)
//
// A (16x32 bf16): lane l holds row (l&15); element e -> K = ((l>=16)?16:0)+e.
//                 => one contiguous 16-element chunk (two 16B loads).
// B (32x16 bf16): lane l holds col (l&15); element e -> K = ((l>=16)?16:0)+e.
//                 => one contiguous 16-element chunk.
// C/D (16x16 f32): lane l holds col (l&15); VGPR r -> row r + ((l>=16)?8:0).
// ---------------------------------------------------------------------------
static __device__ __forceinline__ v16bf concat88(v8bf lo, v8bf hi) {
  v16bf r;
#pragma unroll
  for (int e = 0; e < 8; ++e) { r[e] = lo[e]; r[e + 8] = hi[e]; }
  return r;
}

static __device__ __forceinline__ v16bf load_a_frag(const bf16_t* row, int h8) {
  return concat88(*(const v8bf*)(row + 2 * h8), *(const v8bf*)(row + 2 * h8 + 8));
}

static __device__ __forceinline__ v16bf load16(const bf16_t* p) {
  return concat88(*(const v8bf*)p, *(const v8bf*)(p + 8));
}

static __device__ __forceinline__ v8f wmma_bf16(v16bf a, v16bf b, v8f c) {
  return __builtin_amdgcn_wmma_f32_16x16x32_bf16(
      false, a, false, b, (short)0, c, false, false);
}

// ---------------------------------------------------------------------------
// Cast f32 -> bf16 (vectorized x4)
// ---------------------------------------------------------------------------
__global__ __launch_bounds__(256) void cast_f32_to_bf16_kernel(
    const float* __restrict__ in, bf16_t* __restrict__ out, int n) {
  const int i = (blockIdx.x * 256 + threadIdx.x) * 4;
  if (i + 3 < n) {
    const float4 f = *(const float4*)(in + i);
    v4bf o;
    o[0] = (bf16_t)f.x; o[1] = (bf16_t)f.y;
    o[2] = (bf16_t)f.z; o[3] = (bf16_t)f.w;
    *(v4bf*)(out + i) = o;
  }
}

// ---------------------------------------------------------------------------
// bf16 WMMA GEMM: C[M,N] = A[M,K] * W[K,N] + bias[N]
//   SPLIT=false : write f32 row-major [M,N]   (final output projection)
//   SPLIT=true  : write bf16 head-split layout [B,H,L,DK] (QKV)
// Block tile 128x128, 8 waves (4 M-groups x 2 N-groups), each wave 32x64.
//
// Software pipeline, TRIPLE-buffered LDS with ONE barrier per iteration:
//   iter i: stage(i+1) -> buf (i+1)%3 ; wait TENSORcnt<=1 (in-order => tile i
//   landed) ; barrier ; compute(i) from buf i%3.
// Buffer (i+1)%3 was last read by compute(i-2), which is separated from the
// stage(i+1) writes by barrier(i-1) -> race-free.  TDM DMA (or the fallback
// global->LDS copy) thus overlaps a full iteration of WMMA work.
// ---------------------------------------------------------------------------
template <bool SPLIT>
__global__ __launch_bounds__(256) void gemm_bf16_kernel(
    const bf16_t* __restrict__ A, const bf16_t* __restrict__ W,
    const float* __restrict__ bias, float* __restrict__ outF32,
    bf16_t* __restrict__ outSplit, int M, int N, int K) {
  __shared__ bf16_t As[3][128][40];  // [buf][m][k], rows padded to 80B
  __shared__ bf16_t Bs[3][128][40];  // [buf][n][k] (transposed W)

  const int tid  = threadIdx.x;
  const int wave = tid >> 5;
  const int lane = tid & 31;
  const int half = lane >> 4;
  const int ln   = lane & 15;
  const int waveM = wave & 3;   // 0..3  -> 32-row group
  const int waveN = wave >> 2;  // 0..1  -> 64-col group
  const int mBase = blockIdx.y * 128;
  const int nBase = blockIdx.x * 128;

  // --- staging helpers -----------------------------------------------------
  auto stageA = [&](int buf, int kk) {
#if USE_TDM
    if (wave == 0) {  // wave-uniform; TDM DMAs the whole 128x32 A tile
      tdm_load_tile_a(A + (size_t)mBase * K + kk,
                      (unsigned)(uintptr_t)&As[buf][0][0]);
    }
#else
    const int row = tid >> 1;
    const int c   = (tid & 1) * 16;
    const bf16_t* src = A + (size_t)(mBase + row) * K + kk + c;
    *(v8bf*)&As[buf][row][c]     = *(const v8bf*)(src);
    *(v8bf*)&As[buf][row][c + 8] = *(const v8bf*)(src + 8);
#endif
  };
  auto stageW = [&](int buf, int kk) {  // 32x128 tile, transposed into LDS
    const int kr = tid >> 3;         // 0..31
    const int cc = (tid & 7) * 16;   // 0..112
    const bf16_t* src = W + (size_t)(kk + kr) * N + nBase + cc;
    const v16bf g = *(const v16bf*)src;
#pragma unroll
    for (int j = 0; j < 16; ++j) Bs[buf][cc + j][kr] = g[j];
  };

  v8f acc[2][4] = {};

  // Prologue: stage tile 0 into buffer 0.
  stageA(0, 0);
  stageW(0, 0);

  int buf = 0;
  for (int kk = 0; kk < K; kk += 32) {
    const int nbuf = (buf == 2) ? 0 : buf + 1;
    const bool more = (kk + 32) < K;
    if (more) {  // prefetch next tile while current one is consumed
      stageA(nbuf, kk + 32);
      stageW(nbuf, kk + 32);
    }
#if USE_TDM
    if (wave == 0) {  // in-order TENSORcnt: <=1 means tile kk has landed
      if (more) __builtin_amdgcn_s_wait_tensorcnt(1);
      else      __builtin_amdgcn_s_wait_tensorcnt(0);
    }
#endif
    __syncthreads();  // publish buf (stores + TDM) to all waves

    v16bf af[2], bfr[4];
    af[0] = load_a_frag(&As[buf][waveM * 32 + ln][0],      half * 8);
    af[1] = load_a_frag(&As[buf][waveM * 32 + 16 + ln][0], half * 8);
#pragma unroll
    for (int nt = 0; nt < 4; ++nt)
      bfr[nt] = load16(&Bs[buf][waveN * 64 + nt * 16 + ln][half * 16]);
#pragma unroll
    for (int mt = 0; mt < 2; ++mt)
#pragma unroll
      for (int nt = 0; nt < 4; ++nt)
        acc[mt][nt] = wmma_bf16(af[mt], bfr[nt], acc[mt][nt]);

    buf = nbuf;
  }

  // Epilogue: bias add + store (base pointer + constant stride per row)
#pragma unroll
  for (int mt = 0; mt < 2; ++mt) {
#pragma unroll
    for (int nt = 0; nt < 4; ++nt) {
      const int col = nBase + waveN * 64 + nt * 16 + ln;
      const float bv = bias[col];
      const int m0 = mBase + waveM * 32 + mt * 16 + half * 8;  // rows m0..m0+7
      if (SPLIT) {
        // m -> (b, l), col -> (h, dk); all 128 block rows share the same b.
        const int bb = mBase >> 11;          // / L_
        const int l0 = (m0 & (L_ - 1));
        const int hh = col >> 6;             // / DK_
        const int dk = col & (DK_ - 1);
        bf16_t* dst =
            outSplit + (((size_t)(bb * H_ + hh)) * L_ + l0) * DK_ + dk;
#pragma unroll
        for (int r = 0; r < 8; ++r)
          dst[(size_t)r * DK_] = (bf16_t)(acc[mt][nt][r] + bv);
      } else {
        float* dst = outF32 + (size_t)m0 * N + col;
#pragma unroll
        for (int r = 0; r < 8; ++r)
          dst[(size_t)r * N] = acc[mt][nt][r] + bv;
      }
    }
  }
}

// ---------------------------------------------------------------------------
// Flash attention (causal), one block = (b, h, 128 query rows), 8 waves.
// Each wave owns 16 query rows; loops key chunks of 32; online softmax.
// ---------------------------------------------------------------------------
__global__ __launch_bounds__(256) void attn_kernel(
    const bf16_t* __restrict__ Qp, const bf16_t* __restrict__ Kp,
    const bf16_t* __restrict__ Vp, bf16_t* __restrict__ attnM) {
  __shared__ bf16_t Vt[DK_][40];      // V chunk transposed: [dk][kpos]
  __shared__ bf16_t Pl[8][16][40];    // wave-private probabilities (16x32)

  const int tid  = threadIdx.x;
  const int wave = tid >> 5;
  const int lane = tid & 31;
  const int half = lane >> 4;
  const int ln   = lane & 15;

  const int qBlock = blockIdx.x * 128;
  const int h = blockIdx.y;
  const int b = blockIdx.z;

  const bf16_t* Qh = Qp + ((size_t)(b * H_ + h) * L_) * DK_;
  const bf16_t* Kh = Kp + ((size_t)(b * H_ + h) * L_) * DK_;
  const bf16_t* Vh = Vp + ((size_t)(b * H_ + h) * L_) * DK_;

  const int qBase = qBlock + wave * 16;

  // Q fragments for this wave's 16 rows (A layout), kept in registers.
  const bf16_t* qrow = Qh + (size_t)(qBase + ln) * DK_;
  const v16bf qf0 = load_a_frag(qrow,      half * 8);  // dk 0..31
  const v16bf qf1 = load_a_frag(qrow + 32, half * 8);  // dk 32..63

  float mrun[8], lrun[8];
  v8f o0 = {}, o1 = {}, o2 = {}, o3 = {};
#pragma unroll
  for (int r = 0; r < 8; ++r) { mrun[r] = -1e30f; lrun[r] = 0.0f; }

  const float scale = 0.125f;  // 1/sqrt(64)
  const int kbEnd = qBlock + 128;

  for (int kb = 0; kb < kbEnd; kb += 32) {
    __syncthreads();
    {  // stage V chunk transposed (32 kpos x 64 dk)
      const int kp = tid >> 3;          // 0..31
      const int dc = (tid & 7) * 8;     // 0..56
      const v8bf v = *(const v8bf*)(Vh + (size_t)(kb + kp) * DK_ + dc);
#pragma unroll
      for (int j = 0; j < 8; ++j) Vt[dc + j][kp] = v[j];
    }
    __syncthreads();

    if (kb <= qBase + 15) {  // wave-uniform: EXEC all-ones around WMMA
      if (kb + 32 < kbEnd)
        __builtin_prefetch(Kh + (size_t)(kb + 32 + lane) * DK_, 0, 1);

      // S = Q * K^T for keys [kb, kb+32): two 16-wide column tiles.
      v8f s0 = {}, s1 = {};
      {
        const bf16_t* k0 = Kh + (size_t)(kb + ln) * DK_ + half * 16;
        s0 = wmma_bf16(qf0, *(const v16bf*)(k0),      s0);
        s0 = wmma_bf16(qf1, *(const v16bf*)(k0 + 32), s0);
        const bf16_t* k1 = Kh + (size_t)(kb + 16 + ln) * DK_ + half * 16;
        s1 = wmma_bf16(qf0, *(const v16bf*)(k1),      s1);
        s1 = wmma_bf16(qf1, *(const v16bf*)(k1 + 32), s1);
      }

      // Online softmax update per query row (rows live in 16-lane groups).
#pragma unroll
      for (int r = 0; r < 8; ++r) {
        const int q = qBase + r + half * 8;
        float v0 = ((kb + ln)      <= q) ? s0[r] * scale : -1e30f;
        float v1 = ((kb + 16 + ln) <= q) ? s1[r] * scale : -1e30f;
        float mx = fmaxf(v0, v1);
#pragma unroll
        for (int d = 8; d >= 1; d >>= 1) mx = fmaxf(mx, __shfl_xor(mx, d, 16));
        const float mnew  = fmaxf(mrun[r], mx);
        const float alpha = __expf(mrun[r] - mnew);
        const float p0 = __expf(v0 - mnew);
        const float p1 = __expf(v1 - mnew);
        float rs = p0 + p1;
#pragma unroll
        for (int d = 8; d >= 1; d >>= 1) rs += __shfl_xor(rs, d, 16);
        lrun[r] = lrun[r] * alpha + rs;
        mrun[r] = mnew;
        o0[r] *= alpha; o1[r] *= alpha; o2[r] *= alpha; o3[r] *= alpha;
        Pl[wave][r + half * 8][ln]      = (bf16_t)p0;
        Pl[wave][r + half * 8][16 + ln] = (bf16_t)p1;
      }

      // O += P * V  (A = probs 16x32 from LDS, B = Vt columns)
      const v16bf pf = load_a_frag(&Pl[wave][ln][0], half * 8);
      o0 = wmma_bf16(pf, load16(&Vt[ 0 + ln][half * 16]), o0);
      o1 = wmma_bf16(pf, load16(&Vt[16 + ln][half * 16]), o1);
      o2 = wmma_bf16(pf, load16(&Vt[32 + ln][half * 16]), o2);
      o3 = wmma_bf16(pf, load16(&Vt[48 + ln][half * 16]), o3);
    }
  }

  // Normalize and store merged [B, L, D] bf16 (feeds the output projection).
  float inv[8];
#pragma unroll
  for (int r = 0; r < 8; ++r) inv[r] = 1.0f / lrun[r];
  bf16_t* outBase = attnM + (size_t)b * L_ * D_ + (size_t)h * DK_;
#pragma unroll
  for (int r = 0; r < 8; ++r) {
    const int q = qBase + r + half * 8;
    bf16_t* orow = outBase + (size_t)q * D_;
    orow[ 0 + ln] = (bf16_t)(o0[r] * inv[r]);
    orow[16 + ln] = (bf16_t)(o1[r] * inv[r]);
    orow[32 + ln] = (bf16_t)(o2[r] * inv[r]);
    orow[48 + ln] = (bf16_t)(o3[r] * inv[r]);
  }
}

// ---------------------------------------------------------------------------
// Host-side launcher
// ---------------------------------------------------------------------------
extern "C" void kernel_launch(void* const* d_in, const int* in_sizes, int n_in,
                              void* d_out, int out_size, void* d_ws, size_t ws_size,
                              hipStream_t stream) {
  (void)in_sizes; (void)n_in; (void)out_size; (void)ws_size;

  const float* query = (const float*)d_in[0];
  const float* key   = (const float*)d_in[1];
  const float* value = (const float*)d_in[2];
  // d_in[3] is the causal mask; causality is computed analytically.
  const float* wq = (const float*)d_in[4];
  const float* bq = (const float*)d_in[5];
  const float* wk = (const float*)d_in[6];
  const float* bk = (const float*)d_in[7];
  const float* wv = (const float*)d_in[8];
  const float* bv = (const float*)d_in[9];
  const float* wo = (const float*)d_in[10];
  const float* bo = (const float*)d_in[11];
  float* out = (float*)d_out;

  const size_t nX = (size_t)M_ * D_;  // 4,194,304 activation elements
  const size_t nW = (size_t)D_ * D_;  // 1,048,576 weight elements

  // Workspace carve (256B-aligned chunks), ~64 MiB total.
  char* ws = (char*)d_ws;
  size_t off = 0;
  auto carve = [&](size_t elems) -> bf16_t* {
    bf16_t* p = (bf16_t*)(ws + off);
    off += (elems * sizeof(bf16_t) + 255) & ~(size_t)255;
    return p;
  };
  bf16_t* xq  = carve(nX);  bf16_t* xk  = carve(nX);  bf16_t* xv  = carve(nX);
  bf16_t* wqb = carve(nW);  bf16_t* wkb = carve(nW);
  bf16_t* wvb = carve(nW);  bf16_t* wob = carve(nW);
  bf16_t* Qp  = carve(nX);  bf16_t* Kp  = carve(nX);  bf16_t* Vp  = carve(nX);
  bf16_t* atn = carve(nX);

  auto cast = [&](const float* src, bf16_t* dst, size_t n) {
    const int blocks = (int)((n / 4 + 255) / 256);
    cast_f32_to_bf16_kernel<<<blocks, 256, 0, stream>>>(src, dst, (int)n);
  };
  cast(query, xq, nX);
  cast(key,   xk, nX);
  cast(value, xv, nX);
  cast(wq, wqb, nW);
  cast(wk, wkb, nW);
  cast(wv, wvb, nW);
  cast(wo, wob, nW);

  const dim3 gemmGrid(D_ / 128, M_ / 128);  // (8, 32)
  // QKV projections -> bf16 head-split [B,H,L,DK]
  gemm_bf16_kernel<true><<<gemmGrid, 256, 0, stream>>>(xq, wqb, bq, nullptr, Qp, M_, D_, D_);
  gemm_bf16_kernel<true><<<gemmGrid, 256, 0, stream>>>(xk, wkb, bk, nullptr, Kp, M_, D_, D_);
  gemm_bf16_kernel<true><<<gemmGrid, 256, 0, stream>>>(xv, wvb, bv, nullptr, Vp, M_, D_, D_);

  // Causal flash attention -> merged bf16 [B, L, D]
  attn_kernel<<<dim3(L_ / 128, H_, B_), 256, 0, stream>>>(Qp, Kp, Vp, atn);

  // Output projection -> f32 d_out
  gemm_bf16_kernel<false><<<gemmGrid, 256, 0, stream>>>(atn, wob, bo, out, nullptr, M_, D_, D_);
}
